// DecoderBlock_48679159333606
// MI455X (gfx1250) — compile-verified, matrix operand order checked by reading, not run on hardware
//
#include <hip/hip_runtime.h>
#include <hip/hip_bf16.h>
#include <math.h>

// ---------------------------------------------------------------------------
// CDNA5 / gfx1250 decoder block. All matmul math in bf16 WMMA
// (v_wmma_f32_16x16x32_bf16), fp32 accumulate. wave32 everywhere.
// All WMMA operands are fetched with k-pairs contiguous so the compiler
// vectorizes each fragment into 2x global_load_b128.
// ---------------------------------------------------------------------------

typedef __bf16 bf16;
typedef __attribute__((ext_vector_type(16))) __bf16 v16bf;
typedef __attribute__((ext_vector_type(8)))  float  v8f;
typedef __attribute__((ext_vector_type(2)))  __bf16 bf16x2;

// K-offset of VGPR pair r (0..7) for the CDNA5 16-bit A/B operand layout:
// lanes 0-15: regs 0-3 hold K=0..7, regs 4-7 hold K=16..23
// lanes 16-31: regs 0-3 hold K=8..15, regs 4-7 hold K=24..31
__device__ __forceinline__ int frag_koff(int r, int half) {
    return ((r >> 2) << 4) | (half << 3) | ((r & 3) << 1);
}

// Fragment loader: the 16 "rows" (A rows, or B columns when the source is
// column-major / transposed) striped as lane&15, K-pairs contiguous in memory.
// Element (row, k) = base[row*ld + k0 + k]. Vectorizes to 2x b128 per lane.
__device__ __forceinline__ v16bf load_frag_pairs(const bf16* base, int ld, int k0) {
    const int l = threadIdx.x & 31;
    const int row = l & 15, half = l >> 4;
    const bf16* p = base + (size_t)row * ld + k0;
    v16bf f;
#pragma unroll
    for (int r = 0; r < 8; ++r) {
        const int ko = frag_koff(r, half);
        bf16x2 t = *reinterpret_cast<const bf16x2*>(p + ko);
        f[2 * r]     = t.x;
        f[2 * r + 1] = t.y;
    }
    return f;
}

__device__ __forceinline__ v8f wmma_bf16f32(v16bf a, v16bf b, v8f c) {
    // (neg_a, A, neg_b, B, c_mod, C, reuse_a, reuse_b)
    return __builtin_amdgcn_wmma_f32_16x16x32_bf16(false, a, false, b, (short)0, c,
                                                   false, false);
}

// ---------------------------------------------------------------------------
// fp32 weight (K_in x N, row-major) -> bf16 TRANSPOSED (N x K_out, row-major),
// zero-padding K_out > K_in. B fragments then load with contiguous k-pairs.
// ---------------------------------------------------------------------------
__global__ void convert_T_bf16_kernel(const float* __restrict__ in,
                                      bf16* __restrict__ out,
                                      int K_in, int K_out, int N, int total) {
    int idx = blockIdx.x * blockDim.x + threadIdx.x;
    if (idx >= total) return;
    int n = idx / K_out;
    int k = idx - n * K_out;
    out[idx] = (k < K_in) ? (bf16)in[(size_t)k * N + n] : (bf16)0.0f;
}

// ---------------------------------------------------------------------------
// LayerNorm over last dim -> bf16 (optionally zero-padded to cols_out).
// One wave per row, 8 rows per 256-thread block.
// ---------------------------------------------------------------------------
__global__ __launch_bounds__(256) void layernorm_bf16_kernel(
    const float* __restrict__ X, const float* __restrict__ g,
    const float* __restrict__ b, bf16* __restrict__ out,
    int rows, int cols, int cols_out) {
    const int row = blockIdx.x * 8 + (threadIdx.x >> 5);
    if (row >= rows) return;
    const int l = threadIdx.x & 31;
    const float* x = X + (size_t)row * cols;
    float s = 0.0f, s2 = 0.0f;
    for (int i = l; i < cols; i += 32) {
        float v = x[i];
        s += v;
        s2 += v * v;
    }
#pragma unroll
    for (int m = 16; m >= 1; m >>= 1) {
        s  += __shfl_xor(s, m, 32);
        s2 += __shfl_xor(s2, m, 32);
    }
    const float mean = s / (float)cols;
    const float var  = s2 / (float)cols - mean * mean;
    const float rs   = rsqrtf(var + 1e-5f);
    bf16* o = out + (size_t)row * cols_out;
    for (int i = l; i < cols_out; i += 32) {
        float v = (i < cols) ? (x[i] - mean) * rs * g[i] + b[i] : 0.0f;
        o[i] = (bf16)v;
    }
}

// ---------------------------------------------------------------------------
// GEMM: C[M,N=512] = act(A[M,K]bf16 x B[K,512]bf16 + bias) (+ residual).
// B is supplied TRANSPOSED (Bt[n][k], leading dim ldbt=K_pad) so every operand
// fragment is a pair of b128 loads. No LDS, no barriers; weights are
// L2-resident (<=512 KB).
// Block = 8 waves; per-wave tile 32x64 (2 A frags x 4 B frags = 8 WMMAs per
// k-step, each B fragment reused by both A fragments); block tile 256x64.
// Epilogue options: +bias, exact GELU, +residual->f32, ->bf16, ->bf16
// transposed (outBT[n][m], used to produce channel-major V for attention).
// ---------------------------------------------------------------------------
__global__ __launch_bounds__(256) void gemm_bf16_kernel(
    const bf16* __restrict__ A, int lda, const bf16* __restrict__ Bt, int ldbt,
    int M, int N, int K, const float* __restrict__ bias,
    const float* __restrict__ residual, float* __restrict__ outF,
    bf16* __restrict__ outB, bf16* __restrict__ outBT, int gelu) {
    const int wave = threadIdx.x >> 5;
    const int l = threadIdx.x & 31;
    const int m0 = blockIdx.x * 256 + wave * 32;
    const int n0 = blockIdx.y * 64;
    v8f acc[2][4] = {};
    const bf16* Arow0 = A + (size_t)m0 * lda;
    const bf16* Arow1 = A + (size_t)(m0 + 16) * lda;

    for (int k0 = 0; k0 < K; k0 += 32) {
        v16bf a0 = load_frag_pairs(Arow0, lda, k0);
        v16bf a1 = load_frag_pairs(Arow1, lda, k0);
#pragma unroll
        for (int c = 0; c < 4; ++c) {
            v16bf bb = load_frag_pairs(Bt + (size_t)(n0 + c * 16) * ldbt, ldbt, k0);
            acc[0][c] = wmma_bf16f32(a0, bb, acc[0][c]);
            acc[1][c] = wmma_bf16f32(a1, bb, acc[1][c]);
        }
    }

    const int half = l >> 4, col = l & 15;
#pragma unroll
    for (int t = 0; t < 2; ++t) {
#pragma unroll
        for (int c = 0; c < 4; ++c) {
#pragma unroll
            for (int r = 0; r < 8; ++r) {
                const int m = m0 + t * 16 + half * 8 + r;
                const int n = n0 + c * 16 + col;
                float v = acc[t][c][r];
                if (bias) v += bias[n];
                if (gelu) v = 0.5f * v * (1.0f + erff(v * 0.70710678118654752f));
                if (outF) {
                    const size_t idx = (size_t)m * N + n;
                    outF[idx] = (residual ? residual[idx] : 0.0f) + v;
                }
                if (outB)  outB[(size_t)m * N + n] = (bf16)v;
                if (outBT) outBT[(size_t)n * M + m] = (bf16)v;
            }
        }
    }
}

// ---------------------------------------------------------------------------
// Gated-softmax attention, flash-style online softmax.
// One wave per 16-query tile per (batch, head); 32 keys per iteration.
// Gate: keep softmax numerator only where score > 0 (sigmoid(s)>0.5);
// denominator keeps ALL keys (matches reference: softmax first, then mask).
// Q/K layout: (B, L, H*64) row-major bf16, LD = H*64 = 512.
// V layout: TRANSPOSED channel-major Vt[n][m], n = global channel (h*64+c),
//           m = b*Lk + key, leading dim ld_vt = B*Lk  -> b128 fragment loads.
// ---------------------------------------------------------------------------
__global__ __launch_bounds__(128) void attn_kernel(
    const bf16* __restrict__ Q, const bf16* __restrict__ K,
    const bf16* __restrict__ Vt, bf16* __restrict__ O,
    int Lq, int Lk, int H, int ld_vt, float scale) {
    __shared__ __align__(16) bf16 pbuf[4][16 * 32];  // per-wave gated-P tile
    const int wave = threadIdx.x >> 5;
    const int l = threadIdx.x & 31;
    const int half = l >> 4;
    const int col = l & 15;
    const int q0 = (blockIdx.x * 4 + wave) * 16;
    const int h = blockIdx.y;
    const int b = blockIdx.z;
    const int LD = H * 64;

    const bf16* Qb = Q + ((size_t)b * Lq) * LD + h * 64;
    const bf16* Kb = K + ((size_t)b * Lk) * LD + h * 64;
    const bf16* Vb = Vt + (size_t)(h * 64) * ld_vt + (size_t)b * Lk;

    // Q tile 16x64 lives in registers for the whole key loop (2 A fragments).
    v16bf qa0 = load_frag_pairs(Qb + (size_t)q0 * LD, LD, 0);
    v16bf qa1 = load_frag_pairs(Qb + (size_t)q0 * LD, LD, 32);

    float mstat[8], lstat[8];
#pragma unroll
    for (int r = 0; r < 8; ++r) { mstat[r] = -1e30f; lstat[r] = 0.0f; }
    v8f acc[4] = {};

    for (int j = 0; j < Lk; j += 32) {
        // scores S = (Q K^T) * scale for keys [j, j+32)
        v8f s[2];
#pragma unroll
        for (int t = 0; t < 2; ++t) {
            v16bf kb0 = load_frag_pairs(Kb + (size_t)(j + t * 16) * LD, LD, 0);
            v16bf kb1 = load_frag_pairs(Kb + (size_t)(j + t * 16) * LD, LD, 32);
            v8f c = {};
            c = wmma_bf16f32(qa0, kb0, c);
            c = wmma_bf16f32(qa1, kb1, c);
            s[t] = c;
        }

        float mnew[8], fscale[8];
#pragma unroll
        for (int r = 0; r < 8; ++r) {
            float a0 = s[0][r] * scale;
            float a1 = s[1][r] * scale;
            s[0][r] = a0;
            s[1][r] = a1;
            float t = fmaxf(a0, a1);
#pragma unroll
            for (int msk = 8; msk >= 1; msk >>= 1)
                t = fmaxf(t, __shfl_xor(t, msk, 32));   // row max over 16 cols
            mnew[r] = fmaxf(mstat[r], t);
            fscale[r] = expf(mstat[r] - mnew[r]);
        }

#pragma unroll
        for (int r = 0; r < 8; ++r) {
            float p0 = expf(s[0][r] - mnew[r]);
            float p1 = expf(s[1][r] - mnew[r]);
            float rsum = p0 + p1;
#pragma unroll
            for (int msk = 8; msk >= 1; msk >>= 1)
                rsum += __shfl_xor(rsum, msk, 32);
            lstat[r] = lstat[r] * fscale[r] + rsum;     // ungated denominator
            mstat[r] = mnew[r];
            float g0 = (s[0][r] > 0.0f) ? p0 : 0.0f;    // gated numerator
            float g1 = (s[1][r] > 0.0f) ? p1 : 0.0f;
            const int row = half * 8 + r;
            pbuf[wave][row * 32 + col]      = (bf16)g0;
            pbuf[wave][row * 32 + 16 + col] = (bf16)g1;
        }

#pragma unroll
        for (int c = 0; c < 4; ++c)
#pragma unroll
            for (int r = 0; r < 8; ++r) acc[c][r] *= fscale[r];

        // P (C-layout) -> A-layout via per-wave LDS round trip, then P x V.
        v16bf pa = load_frag_pairs(&pbuf[wave][0], 32, 0);
#pragma unroll
        for (int c = 0; c < 4; ++c) {
            v16bf vbf = load_frag_pairs(Vb + (size_t)(c * 16) * ld_vt + j, ld_vt, 0);
            acc[c] = wmma_bf16f32(pa, vbf, acc[c]);
        }
    }

    bf16* Ob = O + ((size_t)b * Lq) * LD + h * 64;
#pragma unroll
    for (int c = 0; c < 4; ++c)
#pragma unroll
        for (int r = 0; r < 8; ++r) {
            float v = acc[c][r] / lstat[r];
            Ob[(size_t)(q0 + half * 8 + r) * LD + c * 16 + col] = (bf16)v;
        }
}

// ---------------------------------------------------------------------------
// Host-side orchestration. Workspace usage ~64 MB.
// ---------------------------------------------------------------------------
extern "C" void kernel_launch(void* const* d_in, const int* in_sizes, int n_in,
                              void* d_out, int out_size, void* d_ws, size_t ws_size,
                              hipStream_t stream) {
    (void)in_sizes; (void)n_in; (void)out_size; (void)ws_size;

    const float* X    = (const float*)d_in[0];
    const float* Z    = (const float*)d_in[1];
    const float* Wq_s = (const float*)d_in[2];
    const float* Wk_s = (const float*)d_in[3];
    const float* Wv_s = (const float*)d_in[4];
    const float* Wo_s = (const float*)d_in[5];
    const float* Wq_c = (const float*)d_in[6];
    const float* Wk_c = (const float*)d_in[7];
    const float* Wv_c = (const float*)d_in[8];
    const float* Wo_c = (const float*)d_in[9];
    const float* g1   = (const float*)d_in[10];
    const float* b1   = (const float*)d_in[11];
    const float* g2   = (const float*)d_in[12];
    const float* b2   = (const float*)d_in[13];
    const float* W1   = (const float*)d_in[14];
    const float* bf1  = (const float*)d_in[15];
    const float* W2   = (const float*)d_in[16];
    const float* bf2  = (const float*)d_in[17];

    const int BATCH = 4, Lq = 2048, Lz = 1024, D = 512;
    const int MQ = BATCH * Lq;   // 8192 activation rows
    const int MZ = BATCH * Lz;   // 4096 cross K/V rows

    char* wsp = (char*)d_ws;
    auto take = [&](size_t bytes) -> char* {
        char* p = wsp;
        wsp += (bytes + 255) & ~(size_t)255;
        return p;
    };
    float* Xbuf = (float*)take((size_t)MQ * D * 4);   // running residual stream
    bf16* nb  = (bf16*)take((size_t)MQ * D * 2);      // LN output (reused 3x)
    bf16* znb = (bf16*)take((size_t)MZ * 128 * 2);    // LN(Zelta), K-padded 100->128
    bf16* qb  = (bf16*)take((size_t)MQ * D * 2);
    bf16* kb  = (bf16*)take((size_t)MQ * D * 2);
    bf16* vtb = (bf16*)take((size_t)MQ * D * 2);      // V, channel-major
    bf16* ob  = (bf16*)take((size_t)MQ * D * 2);      // attn out / FFN hidden
    // All weights stored transposed: Wt[n][k], leading dim = padded K.
    bf16* wqs = (bf16*)take((size_t)512 * 512 * 2);
    bf16* wks = (bf16*)take((size_t)512 * 512 * 2);
    bf16* wvs = (bf16*)take((size_t)512 * 512 * 2);
    bf16* wos = (bf16*)take((size_t)512 * 512 * 2);
    bf16* wqc = (bf16*)take((size_t)512 * 512 * 2);
    bf16* wkc = (bf16*)take((size_t)512 * 128 * 2);   // zero k 100..127
    bf16* wvc = (bf16*)take((size_t)512 * 128 * 2);
    bf16* woc = (bf16*)take((size_t)512 * 512 * 2);
    bf16* w1b = (bf16*)take((size_t)512 * 512 * 2);
    bf16* w2b = (bf16*)take((size_t)512 * 512 * 2);

    auto convT = [&](const float* src, bf16* dst, int kin, int kout, int n) {
        int total = n * kout;
        convert_T_bf16_kernel<<<dim3((total + 255) / 256), dim3(256), 0, stream>>>(
            src, dst, kin, kout, n, total);
    };
    auto gemm = [&](const bf16* A, int lda, const bf16* Bt, int ldbt, int M, int K,
                    const float* bias, const float* res, float* oF, bf16* oB,
                    bf16* oBT, int gelu) {
        gemm_bf16_kernel<<<dim3(M / 256, 512 / 64), dim3(256), 0, stream>>>(
            A, lda, Bt, ldbt, M, 512, K, bias, res, oF, oB, oBT, gelu);
    };

    // weight conversion: fp32 -> bf16, transposed, cross K/V padded K 100 -> 128
    convT(Wq_s, wqs, 512, 512, 512);
    convT(Wk_s, wks, 512, 512, 512);
    convT(Wv_s, wvs, 512, 512, 512);
    convT(Wo_s, wos, 512, 512, 512);
    convT(Wq_c, wqc, 512, 512, 512);
    convT(Wk_c, wkc, 100, 128, 512);
    convT(Wv_c, wvc, 100, 128, 512);
    convT(Wo_c, woc, 512, 512, 512);
    convT(W1,  w1b, 512, 512, 512);
    convT(W2,  w2b, 512, 512, 512);

    // ---- self attention sublayer ----
    layernorm_bf16_kernel<<<dim3(MQ / 8), dim3(256), 0, stream>>>(X, g1, b1, nb,
                                                                  MQ, 512, 512);
    gemm(nb, 512, wqs, 512, MQ, 512, nullptr, nullptr, nullptr, qb, nullptr, 0);
    gemm(nb, 512, wks, 512, MQ, 512, nullptr, nullptr, nullptr, kb, nullptr, 0);
    gemm(nb, 512, wvs, 512, MQ, 512, nullptr, nullptr, nullptr, nullptr, vtb, 0);
    attn_kernel<<<dim3(Lq / 64, 8, BATCH), dim3(128), 0, stream>>>(
        qb, kb, vtb, ob, Lq, Lq, 8, MQ, 0.125f);
    gemm(ob, 512, wos, 512, MQ, 512, nullptr, X, Xbuf, nullptr, nullptr, 0);

    // ---- cross attention sublayer ----
    layernorm_bf16_kernel<<<dim3(MQ / 8), dim3(256), 0, stream>>>(Xbuf, g1, b1, nb,
                                                                  MQ, 512, 512);
    layernorm_bf16_kernel<<<dim3(MZ / 8), dim3(256), 0, stream>>>(Z, g2, b2, znb,
                                                                  MZ, 100, 128);
    gemm(nb, 512, wqc, 512, MQ, 512, nullptr, nullptr, nullptr, qb, nullptr, 0);
    gemm(znb, 128, wkc, 128, MZ, 128, nullptr, nullptr, nullptr, kb, nullptr, 0);
    gemm(znb, 128, wvc, 128, MZ, 128, nullptr, nullptr, nullptr, nullptr, vtb, 0);
    attn_kernel<<<dim3(Lq / 64, 8, BATCH), dim3(128), 0, stream>>>(
        qb, kb, vtb, ob, Lq, Lz, 8, MZ, 0.125f);
    gemm(ob, 512, woc, 512, MQ, 512, nullptr, Xbuf, Xbuf, nullptr, nullptr, 0);

    // ---- FFN sublayer ----
    layernorm_bf16_kernel<<<dim3(MQ / 8), dim3(256), 0, stream>>>(Xbuf, g1, b1, nb,
                                                                  MQ, 512, 512);
    gemm(nb, 512, w1b, 512, MQ, 512, bf1, nullptr, nullptr, ob, nullptr, 1);
    gemm(ob, 512, w2b, 512, MQ, 512, bf2, Xbuf, (float*)d_out, nullptr, nullptr, 0);
}
